// Generator_Model_3607772528919
// MI455X (gfx1250) — compile-verified, matrix operand order checked by reading, not run on hardware
//
#include <hip/hip_runtime.h>
#include <cstdint>
#include <cstddef>

#ifndef USE_TDM
#define USE_TDM 1
#endif

// ---------------- problem constants ----------------
namespace {
constexpr int Bn = 256, Tn = 20, Vn = 20000, En = 100, Hn = 700;
constexpr int H3 = 2100, DYn = 200, DZn = 500, DECn = 20;
// padded geometry: K-dims to mult of 32, N-dims to mult of 64
constexpr int EP = 128;      // E=100  padded (as K)
constexpr int HP = 704;      // H=700  padded (as K)
constexpr int VP = 20000;    // V as K (already mult 32)
constexpr int H3P = 2112;    // 3H as N padded
constexpr int VNP = 20032;   // V as N padded
constexpr int ENP = 128;     // E as N padded
constexpr float GAMMAf = 0.1f, KEEPf = 0.5f, EPSf = 1e-20f;
}

typedef __attribute__((ext_vector_type(16))) __bf16 v16bf;
typedef __attribute__((ext_vector_type(8)))  float  v8f;
typedef __attribute__((ext_vector_type(4)))  unsigned int v4u;
typedef __attribute__((ext_vector_type(8)))  unsigned int v8u;

__device__ __forceinline__ unsigned short f2bf(float f) {
  union { float f; unsigned int u; } v; v.f = f;
  unsigned int u = v.u;
  unsigned int r = u + 0x7FFFu + ((u >> 16) & 1u);   // round-to-nearest-even
  return (unsigned short)(r >> 16);
}

__device__ __forceinline__ unsigned int hash3(unsigned int a, unsigned int b, unsigned int c) {
  unsigned int h = a * 0x9E3779B1u;
  h ^= b + 0x85EBCA77u + (h << 6) + (h >> 2);
  h ^= c + 0xC2B2AE3Du + (h << 6) + (h >> 2);
  h ^= h >> 16; h *= 0x7FEB352Du; h ^= h >> 15; h *= 0x846CA68Bu; h ^= h >> 16;
  return h;
}

// ---------------- TDM tile staging ----------------
// One Tensor-DMA descriptor: (TKT x 64rows) bf16 tile, row stride ld_elems,
// global -> LDS byte offset lds_off. D# per CDNA5 ISA 8.3/8.4 (2D: groups 2/3 NULL).
#define TM 64
#define TN 64
#define TKT 32

#if USE_TDM
__device__ __forceinline__ void tdm_load_tile(const unsigned short* gsrc, int ld_elems,
                                              unsigned lds_off) {
  unsigned long long ga = (unsigned long long)(size_t)gsrc;
  v4u g0;
  g0[0] = 1u;                                            // count=1 (valid), user mode
  g0[1] = lds_off;                                       // lds_addr (bytes)
  g0[2] = (unsigned)(ga & 0xFFFFFFFFu);                  // global_addr[31:0]
  g0[3] = (unsigned)((ga >> 32) & 0x1FFFFFFu) | (2u << 30); // addr[56:32] | type=2
  unsigned td0 = (unsigned)ld_elems;                     // tensor_dim0 (elements)
  unsigned td1 = 0x100000u;                              // tensor_dim1 (rows, generous)
  unsigned long long s0 = (unsigned long long)(unsigned)ld_elems; // dim0 stride
  v8u g1;
  g1[0] = (1u << 16);                                    // data_size=1 -> 2 bytes
  g1[1] = (td0 & 0xFFFFu) << 16;                         // atomic_addr=0 | td0[15:0]
  g1[2] = (td0 >> 16) | ((td1 & 0xFFFFu) << 16);         // td0[31:16] | td1[15:0]
  g1[3] = ((td1 >> 16) & 0xFFFFu) | ((unsigned)TKT << 16); // td1[31:16] | tile_dim0=32
  g1[4] = (unsigned)TM;                                  // tile_dim1=64 | tile_dim2=0
  g1[5] = (unsigned)(s0 & 0xFFFFFFFFu);                  // dim0_stride[31:0]
  g1[6] = (unsigned)((s0 >> 32) & 0xFFFFu);              // stride[47:32] | d1stride lo=0
  g1[7] = 0u;
  asm volatile("tensor_load_to_lds %0, %1" :: "s"(g0), "s"(g1) : "memory");
}
#endif

// ---------------- bf16 WMMA GEMM ----------------
// C[M,N] = A[M,KP] * B[KP,N] (+bias), B supplied TRANSPOSED: Bt[n][k], n-major.
// K zero-padded to mult 32; M mult 64; Bt has >= ceil(N/64)*64 rows.
// 128 threads = 4 waves; wave w computes tile rows [w*16,w*16+16) x 64 cols (4 WMMAs).
// TDM double-buffered: tile i+1 streams into the ping-pong buffer while tile i computes.
__global__ __launch_bounds__(128) void k_gemm_bf16(
    const unsigned short* __restrict__ A, int lda,
    const unsigned short* __restrict__ Bt, int ldt,
    const float* __restrict__ bias,
    float* __restrict__ C, int ldc,
    int M, int N, int KP)
{
  __shared__ __align__(16) unsigned short As[2][TM * TKT];   // [buf][row][k]
  __shared__ __align__(16) unsigned short Bs[2][TN * TKT];   // [buf][n][k]

  const int tid  = (int)threadIdx.x;
  const int wid  = tid >> 5;
  const int lane = tid & 31;
  const int m0 = (int)blockIdx.y * TM;
  const int n0 = (int)blockIdx.x * TN;

  v8f acc[4] = {};

  const int arow  = (wid << 4) + (lane & 15);
  const int khalf = (lane >> 4) << 3;

#if USE_TDM
  typedef __attribute__((address_space(3))) unsigned short lds_u16;
  const unsigned as_off0 = (unsigned)(unsigned long long)(lds_u16*)As[0];
  const unsigned as_off1 = (unsigned)(unsigned long long)(lds_u16*)As[1];
  const unsigned bs_off0 = (unsigned)(unsigned long long)(lds_u16*)Bs[0];
  const unsigned bs_off1 = (unsigned)(unsigned long long)(lds_u16*)Bs[1];
  // prologue: tile 0 -> buffer 0
  if (tid < 32) {
    tdm_load_tile(A  + (size_t)m0 * lda, lda, as_off0);
    tdm_load_tile(Bt + (size_t)n0 * ldt, ldt, bs_off0);
  }
#endif

  int buf = 0;
  for (int k0 = 0; k0 < KP; k0 += TKT, buf ^= 1) {
#if USE_TDM
    const int knext = k0 + TKT;
    if (knext < KP) {
      if (tid < 32) {       // stream tile i+1 into the other buffer
        tdm_load_tile(A  + (size_t)m0 * lda + knext, lda, buf ? as_off0 : as_off1);
        tdm_load_tile(Bt + (size_t)n0 * ldt + knext, ldt, buf ? bs_off0 : bs_off1);
      }
      __builtin_amdgcn_s_wait_tensorcnt(2);   // tile i's 2 ops done (in-order)
    } else {
      __builtin_amdgcn_s_wait_tensorcnt(0);   // tail: drain everything
    }
#else
    for (int c = tid; c < (TM * TKT) / 8; c += 128) {
      int r = c >> 2, cs = (c & 3) << 3;
      *(uint4*)&As[buf][r * TKT + cs] = *(const uint4*)(A + (size_t)(m0 + r) * lda + k0 + cs);
    }
    for (int c = tid; c < (TN * TKT) / 8; c += 128) {
      int r = c >> 2, cs = (c & 3) << 3;
      *(uint4*)&Bs[buf][r * TKT + cs] = *(const uint4*)(Bt + (size_t)(n0 + r) * ldt + k0 + cs);
    }
#endif
    __syncthreads();                          // tile i visible to all waves

    const unsigned short* asb = As[buf];
    const unsigned short* bsb = Bs[buf];
    union { v16bf v; unsigned int u[8]; } af;
#pragma unroll
    for (int v = 0; v < 8; ++v) {
      int kb = ((v >= 4) ? 16 : 0) + khalf + ((v & 3) << 1);
      af.u[v] = *(const unsigned int*)&asb[arow * TKT + kb];
    }
#pragma unroll
    for (int j = 0; j < 4; ++j) {
      union { v16bf v; unsigned int u[8]; } bfr;
      int col = (j << 4) + (lane & 15);
#pragma unroll
      for (int v = 0; v < 8; ++v) {
        int kb = ((v >= 4) ? 16 : 0) + khalf + ((v & 3) << 1);
        bfr.u[v] = *(const unsigned int*)&bsb[col * TKT + kb];
      }
      acc[j] = __builtin_amdgcn_wmma_f32_16x16x32_bf16(
          false, af.v, false, bfr.v, (short)0, acc[j], false, false);
    }
    __syncthreads();                          // all reads of buf done before overwrite
  }

  // D layout: lane l, VGPR v -> M = v + (l>=16?8:0), N = l%16. M always in-bounds.
  const int rb = m0 + (wid << 4) + ((lane >> 4) << 3);
  const int cb = lane & 15;
#pragma unroll
  for (int j = 0; j < 4; ++j) {
    int gn = n0 + (j << 4) + cb;
    if (gn >= N) continue;
    float badd = bias ? bias[gn] : 0.0f;
#pragma unroll
    for (int v = 0; v < 8; ++v)
      C[(size_t)(rb + v) * ldc + gn] = acc[j][v] + badd;
  }
}

// ---------------- conversion / padding kernels ----------------
__global__ __launch_bounds__(256) void k_cvtT_bf16(const float* __restrict__ src,
                                                   unsigned short* __restrict__ dst,
                                                   int K, int N, int KP, long long total) {
  long long i = (long long)blockIdx.x * 256 + threadIdx.x;
  if (i >= total) return;
  int r = (int)(i / KP);
  int c = (int)(i - (long long)r * KP);
  float v = (r < N && c < K) ? src[(size_t)c * N + r] : 0.0f;
  dst[i] = f2bf(v);
}

__global__ __launch_bounds__(256) void k_cvtP_bf16(const float* __restrict__ src,
                                                   unsigned short* __restrict__ dst,
                                                   int K, int KP, long long total) {
  long long i = (long long)blockIdx.x * 256 + threadIdx.x;
  if (i >= total) return;
  int r = (int)(i / KP);
  int c = (int)(i - (long long)r * KP);
  dst[i] = (c < K) ? f2bf(src[(size_t)r * K + c]) : (unsigned short)0;
}

// ---------------- model kernels ----------------
__global__ __launch_bounds__(256) void k_init_h(
    const int* __restrict__ labels, const float* __restrict__ z,
    const float* __restrict__ W1, const float* __restrict__ b1,
    float* __restrict__ h_ori, float* __restrict__ h_dec,
    float* __restrict__ teach)
{
  int b = (int)blockIdx.x;
  float lab = (float)labels[b];
  for (int j = (int)threadIdx.x; j < Hn; j += 256) {
    float vo, vt;
    if (j < DYn) {
      vo = lab * W1[j] + b1[j];
      vt = (1.0f - lab) * W1[j] + b1[j];
    } else {
      float zz = z[(size_t)b * DZn + (j - DYn)];
      vo = zz; vt = zz;
    }
    h_ori[(size_t)b * Hn + j] = vo;
    h_dec[(size_t)b * Hn + j] = vt;
    teach[(size_t)b * (Tn + 1) * Hn + j] = vo;        // teach_h[:,0,:]
  }
}

__global__ __launch_bounds__(128) void k_gather_emb(
    const int* __restrict__ dec_inputs, const float* __restrict__ emb,
    unsigned short* __restrict__ decemb_bf, unsigned short* __restrict__ inp_bf)
{
  int row = (int)blockIdx.x;          // b*T + t
  int tok = dec_inputs[row];
  int t = row % Tn;
  int b = row / Tn;
  for (int j = (int)threadIdx.x; j < EP; j += 128) {
    unsigned short bf = (j < En) ? f2bf(emb[(size_t)tok * En + j]) : (unsigned short)0;
    decemb_bf[(size_t)row * EP + j] = bf;
    if (t == 0) inp_bf[(size_t)b * EP + j] = bf;      // go = dec_emb[:,0,:]
  }
}

__device__ __forceinline__ float sigm(float x) { return 1.0f / (1.0f + __expf(-x)); }

__global__ __launch_bounds__(256) void k_gru_combine(
    const float* __restrict__ XG, const float* __restrict__ HG,
    float* __restrict__ h, float* __restrict__ teach, int t)
{
  int b = (int)blockIdx.x;
  size_t xr = ((size_t)b * Tn + t) * H3;
  size_t hr = (size_t)b * H3;
  for (int j = (int)threadIdx.x; j < Hn; j += 256) {
    float xz = XG[xr + j], xg_r = XG[xr + Hn + j], xh = XG[xr + 2 * Hn + j];
    float hz = HG[hr + j], hg_r = HG[hr + Hn + j], hh = HG[hr + 2 * Hn + j];
    float zg = sigm(xz + hz);
    float rg = sigm(xg_r + hg_r);
    float cand = tanhf(xh + rg * hh);
    float hp = h[(size_t)b * Hn + j];
    float hn = zg * hp + (1.0f - zg) * cand;
    h[(size_t)b * Hn + j] = hn;
    teach[((size_t)b * (Tn + 1) + (t + 1)) * Hn + j] = hn;
  }
}

__global__ __launch_bounds__(256) void k_drop_g(const float* __restrict__ teach,
                                                unsigned short* __restrict__ Ag) {
  int i = (int)blockIdx.x * 256 + (int)threadIdx.x;
  const int n = Bn * Tn * HP;
  if (i >= n) return;
  int row = i / HP, j = i - row * HP;
  if (j >= Hn) { Ag[i] = 0; return; }
  int b = row / Tn, t = row - b * Tn;
  float v = teach[((size_t)b * (Tn + 1) + (t + 1)) * Hn + j];
  unsigned int h = hash3(0x5EED0001u, (unsigned)row, (unsigned)j);
  Ag[i] = f2bf((h & 1u) ? v * (1.0f / KEEPf) : 0.0f);
}

__global__ __launch_bounds__(256) void k_dstep_combine(
    const float* __restrict__ XG, const float* __restrict__ HG,
    float* __restrict__ h, float* __restrict__ soft,
    unsigned short* __restrict__ out_bf, int t)
{
  int b = (int)blockIdx.x;
  size_t r3 = (size_t)b * H3;
  for (int j = (int)threadIdx.x; j < HP; j += 256) {
    if (j >= Hn) { out_bf[(size_t)b * HP + j] = 0; continue; }
    float hp = h[(size_t)b * Hn + j];
    soft[((size_t)b * DECn + t) * Hn + j] = hp;       // h_prev collected
    float xz = XG[r3 + j], xr = XG[r3 + Hn + j], xh = XG[r3 + 2 * Hn + j];
    float hz = HG[r3 + j], hr = HG[r3 + Hn + j], hh = HG[r3 + 2 * Hn + j];
    float zg = sigm(xz + hz);
    float rg = sigm(xr + hr);
    float cand = tanhf(xh + rg * hh);
    float hn = zg * hp + (1.0f - zg) * cand;
    h[(size_t)b * Hn + j] = hn;
    unsigned int m = hash3(0xD20F0000u + (unsigned)t, (unsigned)b, (unsigned)j);
    out_bf[(size_t)b * HP + j] = f2bf((m & 1u) ? hn * (1.0f / KEEPf) : 0.0f);
  }
}

__device__ __forceinline__ float gumbel_x(const float* logits, size_t base, int t, int b, int v) {
  unsigned int h = hash3(0xA11CE000u + (unsigned)t, (unsigned)b, (unsigned)v);
  float U = (float)(h >> 8) * 5.9604645e-08f;          // [0,1)
  float G = -__logf(-__logf(U + EPSf) + EPSf);
  return (logits[base + v] + G) * (1.0f / GAMMAf);
}

__global__ __launch_bounds__(256) void k_gumbel_softmax(
    const float* __restrict__ logits, unsigned short* __restrict__ prob_bf, int t)
{
  __shared__ float sb[256];
  int b = (int)blockIdx.x, tid = (int)threadIdx.x;
  size_t base = (size_t)b * Vn;

  float mx = -3.4e38f;
  for (int v = tid; v < Vn; v += 256) mx = fmaxf(mx, gumbel_x(logits, base, t, b, v));
  sb[tid] = mx; __syncthreads();
  for (int s = 128; s > 0; s >>= 1) { if (tid < s) sb[tid] = fmaxf(sb[tid], sb[tid + s]); __syncthreads(); }
  mx = sb[0]; __syncthreads();

  float sum = 0.0f;
  for (int v = tid; v < Vn; v += 256) sum += __expf(gumbel_x(logits, base, t, b, v) - mx);
  sb[tid] = sum; __syncthreads();
  for (int s = 128; s > 0; s >>= 1) { if (tid < s) sb[tid] += sb[tid + s]; __syncthreads(); }
  float inv = 1.0f / sb[0];

  for (int v = tid; v < Vn; v += 256)
    prob_bf[base + v] = f2bf(__expf(gumbel_x(logits, base, t, b, v) - mx) * inv);
}

// ---------------- host launcher ----------------
extern "C" void kernel_launch(void* const* d_in, const int* in_sizes, int n_in,
                              void* d_out, int out_size, void* d_ws, size_t ws_size,
                              hipStream_t stream) {
  (void)in_sizes; (void)n_in; (void)out_size; (void)ws_size;
  const int*   labels = (const int*)  d_in[0];
  const float* z      = (const float*)d_in[1];
  const int*   dec_in = (const int*)  d_in[2];
  const float* emb    = (const float*)d_in[3];
  const float* W1     = (const float*)d_in[4];
  const float* b1     = (const float*)d_in[5];
  const float* gru_k  = (const float*)d_in[6];
  const float* gru_r  = (const float*)d_in[7];
  const float* gru_b  = (const float*)d_in[8];
  const float* cell_k = (const float*)d_in[9];
  const float* cell_r = (const float*)d_in[10];
  const float* cell_b = (const float*)d_in[11];
  const float* Wp     = (const float*)d_in[12];
  const float* bp     = (const float*)d_in[13];

  float* out   = (float*)d_out;
  float* teach = out;                                       // B x (T+1) x H
  float* glog  = teach + (size_t)Bn * (Tn + 1) * Hn;        // (B*T) x V
  float* soft  = glog + (size_t)Bn * Tn * Vn;               // B x DEC x H

  char* ws = (char*)d_ws;
  size_t off = 0;
  auto alloc = [&](size_t bytes) -> void* {
    void* p = ws + off;
    off = (off + bytes + 255) & ~(size_t)255;
    return p;
  };
  // transposed, padded bf16 weights (Bt: Nrows x Kcols)
  unsigned short* gruK_T  = (unsigned short*)alloc((size_t)H3P * EP * 2);
  unsigned short* gruR_T  = (unsigned short*)alloc((size_t)H3P * HP * 2);
  unsigned short* cellK_T = (unsigned short*)alloc((size_t)H3P * EP * 2);
  unsigned short* cellR_T = (unsigned short*)alloc((size_t)H3P * HP * 2);
  unsigned short* Wp_T    = (unsigned short*)alloc((size_t)VNP * HP * 2);
  unsigned short* emb_T   = (unsigned short*)alloc((size_t)ENP * VP * 2);
  // padded bf16 activations
  unsigned short* decemb_bf = (unsigned short*)alloc((size_t)Bn * Tn * EP * 2);
  unsigned short* h_bf      = (unsigned short*)alloc((size_t)Bn * HP * 2);
  unsigned short* Ag_bf     = (unsigned short*)alloc((size_t)Bn * Tn * HP * 2);
  unsigned short* out_bf    = (unsigned short*)alloc((size_t)Bn * HP * 2);
  unsigned short* prob_bf   = (unsigned short*)alloc((size_t)Bn * VP * 2);
  unsigned short* inp_bf    = (unsigned short*)alloc((size_t)Bn * EP * 2);
  // fp32 intermediates
  float* XG      = (float*)alloc((size_t)Bn * Tn * H3 * 4);
  float* HG      = (float*)alloc((size_t)Bn * H3 * 4);
  float* h_teach = (float*)alloc((size_t)Bn * Hn * 4);
  float* h_dec   = (float*)alloc((size_t)Bn * Hn * 4);
  float* XGd     = (float*)alloc((size_t)Bn * H3 * 4);
  float* HGd     = (float*)alloc((size_t)Bn * H3 * 4);
  float* logitsW = (float*)alloc((size_t)Bn * Vn * 4);
  float* inp_f32 = (float*)alloc((size_t)Bn * En * 4);

  auto cvtT = [&](const float* s, unsigned short* d, int K, int N, int KP, int NPr) {
    long long tot = (long long)NPr * KP;
    k_cvtT_bf16<<<(unsigned)((tot + 255) / 256), 256, 0, stream>>>(s, d, K, N, KP, tot);
  };
  auto cvtP = [&](const float* s, unsigned short* d, int rows, int K, int KP) {
    long long tot = (long long)rows * KP;
    k_cvtP_bf16<<<(unsigned)((tot + 255) / 256), 256, 0, stream>>>(s, d, K, KP, tot);
  };
  auto gemm = [&](const unsigned short* A, int lda, const unsigned short* Bt, int ldt,
                  const float* bias, float* C, int ldc, int M, int N, int KP) {
    dim3 g((unsigned)((N + TN - 1) / TN), (unsigned)((M + TM - 1) / TM));
    k_gemm_bf16<<<g, 128, 0, stream>>>(A, lda, Bt, ldt, bias, C, ldc, M, N, KP);
  };

  // weight quantization: bf16, transposed, padded (stays resident in 192MB L2)
  cvtT(gru_k,  gruK_T,  En, H3, EP, H3P);
  cvtT(gru_r,  gruR_T,  Hn, H3, HP, H3P);
  cvtT(cell_k, cellK_T, En, H3, EP, H3P);
  cvtT(cell_r, cellR_T, Hn, H3, HP, H3P);
  cvtT(Wp,     Wp_T,    Hn, Vn, HP, VNP);
  cvtT(emb,    emb_T,   Vn, En, VP, ENP);

  k_init_h<<<Bn, 256, 0, stream>>>(labels, z, W1, b1, h_teach, h_dec, teach);
  k_gather_emb<<<Bn * Tn, 128, 0, stream>>>(dec_in, emb, decemb_bf, inp_bf);

  // hoisted x-gates for the whole teacher sequence: XG = dec_emb @ gru_k + b0
  gemm(decemb_bf, EP, gruK_T, EP, gru_b, XG, H3, Bn * Tn, H3, EP);

  // teacher-forced GRU scan
  for (int t = 0; t < Tn; ++t) {
    cvtP(h_teach, h_bf, Bn, Hn, HP);
    gemm(h_bf, HP, gruR_T, HP, gru_b + H3, HG, H3, Bn, H3, HP);
    k_gru_combine<<<Bn, 256, 0, stream>>>(XG, HG, h_teach, teach, t);
  }

  // dropout + projection: g_logits = drop(g_flat) @ Wp + bp   (143 GFLOP, WMMA-bound)
  k_drop_g<<<(Bn * Tn * HP + 255) / 256, 256, 0, stream>>>(teach, Ag_bf);
  gemm(Ag_bf, HP, Wp_T, HP, bp, glog, Vn, Bn * Tn, Vn, HP);

  // free-running decoder with Gumbel-softmax feedback
  for (int t = 0; t < DECn; ++t) {
    cvtP(h_dec, h_bf, Bn, Hn, HP);
    gemm(inp_bf, EP, cellK_T, EP, cell_b,      XGd, H3, Bn, H3, EP);
    gemm(h_bf,  HP, cellR_T, HP, cell_b + H3, HGd, H3, Bn, H3, HP);
    k_dstep_combine<<<Bn, 256, 0, stream>>>(XGd, HGd, h_dec, soft, out_bf, t);
    gemm(out_bf, HP, Wp_T, HP, bp, logitsW, Vn, Bn, Vn, HP);
    k_gumbel_softmax<<<Bn, 256, 0, stream>>>(logitsW, prob_bf, t);
    gemm(prob_bf, VP, emb_T, VP, nullptr, inp_f32, En, Bn, En, VP);
    cvtP(inp_f32, inp_bf, Bn, En, EP);
  }
}
